// LSTMAutoencoder_62062277427384
// MI455X (gfx1250) — matrix operand order repaired and checked by reading, not compiled, not run on hardware
//
#include <hip/hip_runtime.h>
#include <hip/hip_bf16.h>

// ---------------------------------------------------------------------------
// LSTM autoencoder for MI455X (gfx1250, wave32).
// B=256, T=256, D=128, H=256.
// Strategy: bf16 WMMA (v_wmma_f32_16x16x32_bf16) with fp32 accumulate.
// One wave computes a 16x16 (batch x hidden-unit) tile for ALL FOUR gates,
// then applies the gate nonlinearities + cell update in-register.
// ---------------------------------------------------------------------------

typedef __bf16 bf16_t;
typedef __attribute__((ext_vector_type(16))) __bf16 v16bf;
typedef __attribute__((ext_vector_type(8)))  __bf16 v8bf;
typedef __attribute__((ext_vector_type(8)))  float  v8f;

#define B_ 256
#define T_ 256
#define D_ 128
#define H_ 256

__device__ __forceinline__ float sigmoidf_(float x) {
    return 1.0f / (1.0f + __expf(-x));
}

// ---------------- prep kernels -------------------------------------------

__global__ void cvt_f32_bf16_kernel(const float* __restrict__ src,
                                    bf16_t* __restrict__ dst, int n) {
    for (int i = blockIdx.x * blockDim.x + threadIdx.x; i < n;
         i += gridDim.x * blockDim.x)
        dst[i] = (bf16_t)src[i];
}

__global__ void add_bias_kernel(const float* __restrict__ a,
                                const float* __restrict__ b,
                                float* __restrict__ dst, int n) {
    for (int i = blockIdx.x * blockDim.x + threadIdx.x; i < n;
         i += gridDim.x * blockDim.x)
        dst[i] = a[i] + b[i];
}

__global__ void zero_u32_kernel(unsigned* __restrict__ p, int n) {
    for (int i = blockIdx.x * blockDim.x + threadIdx.x; i < n;
         i += gridDim.x * blockDim.x)
        p[i] = 0u;
}

// ---------------- fused LSTM step ----------------------------------------
// gates[b, q*HOUT + n] = bsum[...] + x(t) . Wih^T + h_prev . Whh^T
// One wave -> 16 batch rows x 16 hidden cols, 4 gate accumulators.
// A-frag (16x32 bf16): lane l: row = l&15; elems[j]=K(k0+half*16+j).
// B-frag (32x16 bf16): lane l: col = l&15; elems[j]=K(k0+half*16+j).
// C/D f32: lane l: N=l&15, VGPR v: M=v+8*half.
template <int DIN, int HOUT>
__global__ void lstm_step_kernel(
    const bf16_t* __restrict__ xseq,   // [B_, T_, DIN] bf16
    const bf16_t* __restrict__ Wih,    // [4*HOUT, DIN] bf16 row-major
    const bf16_t* __restrict__ Whh,    // [4*HOUT, HOUT] bf16 row-major
    const float*  __restrict__ bsum,   // [4*HOUT] = bih + bhh
    const bf16_t* __restrict__ h_in,   // [B_, HOUT]
    bf16_t*       __restrict__ h_out,  // [B_, HOUT]
    float*        __restrict__ c,      // [B_, HOUT] fp32 cell state
    bf16_t*       __restrict__ seq_bf, // [B_, T_, HOUT] or nullptr
    float*        __restrict__ seq_f,  // [B_, T_, HOUT] or nullptr
    int t)
{
    constexpr int NT = HOUT / 16;                   // hidden tiles
    const int lane = threadIdx.x & 31;
    const int wave = threadIdx.x >> 5;
    const int tile = blockIdx.x * (blockDim.x >> 5) + wave;
    const int tm = tile / NT;                       // batch tile
    const int tn = tile % NT;                       // hidden tile
    const int hw = lane >> 4;                       // half-wave (0/1)
    const int lr = lane & 15;

    const int arow = tm * 16 + lr;                  // A-matrix row (batch idx)
    const int ncol = tn * 16 + lr;                  // N column (hidden idx)

    // init accumulators with combined bias (broadcast along M)
    v8f acc[4];
#pragma unroll
    for (int q = 0; q < 4; ++q) {
        const float bv = bsum[q * HOUT + ncol];
#pragma unroll
        for (int v = 0; v < 8; ++v) acc[q][v] = bv;
    }

    // ---- input contribution: A from xseq[arow, t, :], K over DIN --------
    const bf16_t* xrow = xseq + ((size_t)arow * T_ + t) * DIN;
#pragma unroll
    for (int k0 = 0; k0 < DIN; k0 += 32) {
        const v16bf a = *(const v16bf*)(xrow + k0 + hw * 16);
#pragma unroll
        for (int q = 0; q < 4; ++q) {
            const v16bf b = *(const v16bf*)(
                Wih + (size_t)(q * HOUT + ncol) * DIN + k0 + hw * 16);
            acc[q] = __builtin_amdgcn_wmma_f32_16x16x32_bf16(
                false, a, false, b, (short)0, acc[q], false, false);
        }
    }

    // prefetch next timestep's input row (global_prefetch_b8)
    if (t + 1 < T_) __builtin_prefetch((const void*)(xrow + DIN), 0, 1);

    // ---- recurrent contribution: A from h_in[arow, :], K over HOUT ------
    const bf16_t* hrow = h_in + (size_t)arow * HOUT;
#pragma unroll
    for (int k0 = 0; k0 < HOUT; k0 += 32) {
        const v16bf a = *(const v16bf*)(hrow + k0 + hw * 16);
#pragma unroll
        for (int q = 0; q < 4; ++q) {
            const v16bf b = *(const v16bf*)(
                Whh + (size_t)(q * HOUT + ncol) * HOUT + k0 + hw * 16);
            acc[q] = __builtin_amdgcn_wmma_f32_16x16x32_bf16(
                false, a, false, b, (short)0, acc[q], false, false);
        }
    }

    // ---- gate nonlinearities + state update (torch order i,f,g,o) -------
#pragma unroll
    for (int v = 0; v < 8; ++v) {
        const int brow = tm * 16 + v + 8 * hw;      // C/D layout: M=v+8*half
        const float iv = sigmoidf_(acc[0][v]);
        const float fv = sigmoidf_(acc[1][v]);
        const float gv = tanhf(acc[2][v]);
        const float ov = sigmoidf_(acc[3][v]);
        const size_t idx = (size_t)brow * HOUT + ncol;
        const float cn = fv * c[idx] + iv * gv;
        c[idx] = cn;
        const float hn = ov * tanhf(cn);
        h_out[idx] = (bf16_t)hn;
        const size_t sidx = ((size_t)brow * T_ + t) * HOUT + ncol;
        if (seq_bf) seq_bf[sidx] = (bf16_t)hn;
        if (seq_f)  seq_f[sidx]  = hn;
    }
}

// ---------------- host driver --------------------------------------------

extern "C" void kernel_launch(void* const* d_in, const int* in_sizes, int n_in,
                              void* d_out, int out_size, void* d_ws, size_t ws_size,
                              hipStream_t stream) {
    (void)in_sizes; (void)n_in; (void)out_size; (void)ws_size;

    const float* x       = (const float*)d_in[0];   // [B,T,D]
    const float* encWih  = (const float*)d_in[1];   // [4H,D]
    const float* encWhh  = (const float*)d_in[2];   // [4H,H]
    const float* encBih  = (const float*)d_in[3];
    const float* encBhh  = (const float*)d_in[4];
    const float* decWih  = (const float*)d_in[5];   // [4D,H]
    const float* decWhh  = (const float*)d_in[6];   // [4D,D]
    const float* decBih  = (const float*)d_in[7];
    const float* decBhh  = (const float*)d_in[8];
    float* out = (float*)d_out;                     // [B,T,D] fp32

    // carve workspace (256B aligned slices); total ~50 MB
    char* w = (char*)d_ws;
    auto carve = [&](size_t bytes) -> char* {
        char* p = w; w += (bytes + 255) & ~(size_t)255; return p;
    };
    bf16_t* x_bf      = (bf16_t*)carve((size_t)B_ * T_ * D_ * 2);   // 16 MB
    bf16_t* encoded   = (bf16_t*)carve((size_t)B_ * T_ * H_ * 2);   // 32 MB
    bf16_t* eWih_bf   = (bf16_t*)carve((size_t)4 * H_ * D_ * 2);
    bf16_t* eWhh_bf   = (bf16_t*)carve((size_t)4 * H_ * H_ * 2);
    bf16_t* dWih_bf   = (bf16_t*)carve((size_t)4 * D_ * H_ * 2);
    bf16_t* dWhh_bf   = (bf16_t*)carve((size_t)4 * D_ * D_ * 2);
    float*  bsum_enc  = (float*)carve((size_t)4 * H_ * 4);
    float*  bsum_dec  = (float*)carve((size_t)4 * D_ * 4);
    bf16_t* enc_h0    = (bf16_t*)carve((size_t)B_ * H_ * 2);
    bf16_t* enc_h1    = (bf16_t*)carve((size_t)B_ * H_ * 2);
    float*  enc_c     = (float*) carve((size_t)B_ * H_ * 4);
    bf16_t* dec_h0    = (bf16_t*)carve((size_t)B_ * D_ * 2);
    bf16_t* dec_h1    = (bf16_t*)carve((size_t)B_ * D_ * 2);
    float*  dec_c     = (float*) carve((size_t)B_ * D_ * 4);

    // ---- prep: fp32 -> bf16 conversions, bias sums, zero states ----
    cvt_f32_bf16_kernel<<<2048, 256, 0, stream>>>(x, x_bf, B_ * T_ * D_);
    cvt_f32_bf16_kernel<<<256, 256, 0, stream>>>(encWih, eWih_bf, 4 * H_ * D_);
    cvt_f32_bf16_kernel<<<256, 256, 0, stream>>>(encWhh, eWhh_bf, 4 * H_ * H_);
    cvt_f32_bf16_kernel<<<256, 256, 0, stream>>>(decWih, dWih_bf, 4 * D_ * H_);
    cvt_f32_bf16_kernel<<<256, 256, 0, stream>>>(decWhh, dWhh_bf, 4 * D_ * D_);
    add_bias_kernel<<<4, 256, 0, stream>>>(encBih, encBhh, bsum_enc, 4 * H_);
    add_bias_kernel<<<2, 256, 0, stream>>>(decBih, decBhh, bsum_dec, 4 * D_);
    zero_u32_kernel<<<64, 256, 0, stream>>>((unsigned*)enc_h0, B_ * H_ / 2);
    zero_u32_kernel<<<64, 256, 0, stream>>>((unsigned*)enc_c, B_ * H_);
    zero_u32_kernel<<<64, 256, 0, stream>>>((unsigned*)dec_h0, B_ * D_ / 2);
    zero_u32_kernel<<<64, 256, 0, stream>>>((unsigned*)dec_c, B_ * D_);

    bf16_t* ench[2] = { enc_h0, enc_h1 };
    bf16_t* dech[2] = { dec_h0, dec_h1 };

    // ---- encoder: 256 tiles (16 batch x 16 hidden), 4 waves/block ----
    const int encTiles = (B_ / 16) * (H_ / 16);   // 256
    for (int t = 0; t < T_; ++t) {
        lstm_step_kernel<D_, H_><<<encTiles / 4, 128, 0, stream>>>(
            x_bf, eWih_bf, eWhh_bf, bsum_enc,
            ench[t & 1], ench[(t + 1) & 1], enc_c,
            encoded, nullptr, t);
    }

    // ---- decoder: 128 tiles, output fp32 straight to d_out ----
    const int decTiles = (B_ / 16) * (D_ / 16);   // 128
    for (int t = 0; t < T_; ++t) {
        lstm_step_kernel<H_, D_><<<decTiles / 4, 128, 0, stream>>>(
            encoded, dWih_bf, dWhh_bf, bsum_dec,
            dech[t & 1], dech[(t + 1) & 1], dec_c,
            nullptr, out, t);
    }
}
